// single_head_attention_89103391523398
// MI455X (gfx1250) — matrix operand order repaired and checked by reading, not compiled, not run on hardware
//
#include <hip/hip_runtime.h>
#include <math.h>

typedef __attribute__((ext_vector_type(16))) _Float16 v16h;
typedef __attribute__((ext_vector_type(8)))  _Float16 v8h;
typedef __attribute__((ext_vector_type(8)))  float    v8f;
typedef __attribute__((ext_vector_type(4)))  unsigned v4u;
typedef __attribute__((ext_vector_type(8)))  int      v8i;
typedef __attribute__((ext_vector_type(4)))  int      v4i;

#define SEQ   8192
#define D_IN  512
#define D_OUT 64

#if defined(__has_builtin)
#if __has_builtin(__builtin_amdgcn_tensor_load_to_lds) && \
    __has_builtin(__builtin_amdgcn_s_wait_tensorcnt)
#define HAVE_TDM 1
#endif
#endif

// CDNA5 wave32 WMMA f16 layouts (ISA 7.12.2):
//  A (16x32): lane L -> M=L&15 ; K = 16*hi+[0..15] (this code's order, not the ISA table)
//  B (32x16): lane L -> N=L&15 ; K = 16*hi+[0..15] (same code order as A)
//  C (16x16 f32): lane L, slot r -> M = r+8*hi, N = L&15
// => A-frag = two contiguous 16B loads from (M,K)-major; B-frag = one 32B load
//    from (N,K)-major storage.

__device__ __forceinline__ v16h cat8(v8h lo, v8h hi) {
  return __builtin_shufflevector(lo, hi, 0, 1, 2, 3, 4, 5, 6, 7,
                                 8, 9, 10, 11, 12, 13, 14, 15);
}
__device__ __forceinline__ v8f wmma16(v16h a, v16h b, v8f c) {
  return __builtin_amdgcn_wmma_f32_16x16x32_f16(false, a, false, b,
                                                (short)0, c, false, false);
}
__device__ __forceinline__ v16h load_afrag(const _Float16* base, int hi) {
  v8h lo = *(const v8h*)(base + 16 * hi);
  v8h hh = *(const v8h*)(base + 16 * hi + 8);
  return cat8(lo, hh);
}

#ifdef HAVE_TDM
// Issue a TDM 2D tile load (f16 elements) global -> LDS.  D# per ISA 8.3/8.4:
//  group0: [1:0]=count=1, [63:32]=lds byte addr, [120:64]=tile global addr,
//          [127:126]=type=2.
//  group1: data_size=1 (2B), tensor_dim0/1, tile_dim0/1, tensor_dim0_stride.
__device__ __forceinline__ void tdm_load_2d(unsigned lds_addr, const void* gptr,
                                            unsigned t_d0, unsigned t_d1,
                                            unsigned tile_d0, unsigned tile_d1,
                                            unsigned stride0) {
  unsigned long long ga = (unsigned long long)gptr;
  v4u g0;
  g0.x = 1u;                                            // count=1, user D#
  g0.y = lds_addr;                                      // LDS dest (bytes)
  g0.z = (unsigned)ga;                                  // global_addr[31:0]
  g0.w = (unsigned)((ga >> 32) & 0x01FFFFFFu) | (2u << 30);  // [56:32]|type=2
  v8i g1;
  g1[0] = (int)(1u << 16);                              // data_size = 2 bytes
  g1[1] = (int)((t_d0 & 0xFFFFu) << 16);                // tensor_dim0[15:0]
  g1[2] = (int)(((t_d0 >> 16) & 0xFFFFu) | ((t_d1 & 0xFFFFu) << 16));
  g1[3] = (int)(((t_d1 >> 16) & 0xFFFFu) | ((tile_d0 & 0xFFFFu) << 16));
  g1[4] = (int)(tile_d1 & 0xFFFFu);                     // tile_dim1, tile_dim2=0
  g1[5] = (int)stride0;                                 // dim0 stride [31:0]
  g1[6] = 0;                                            // stride hi, dim1 stride
  g1[7] = 0;
  v4i z4 = {0, 0, 0, 0};
#if __clang_major__ >= 23
  v8i z8 = {0, 0, 0, 0, 0, 0, 0, 0};
  __builtin_amdgcn_tensor_load_to_lds(g0, g1, z4, z4, z8, 0);
#else
  __builtin_amdgcn_tensor_load_to_lds(g0, g1, z4, z4, 0);
#endif
}
__device__ __forceinline__ unsigned lds_off(const void* p) {
  return (unsigned)(unsigned long long)p;
}
#endif

// =====================================================================
// Kernel 1: projections  Yd = M (64x512) @ X (512x8192), f16 outputs.
//  Qd -> (seq,64)       Kd -> (seq,64) * 1/8       Vd -> (64,seq)
// grid=(SEQ/64, 3), block=256 (8 waves); WG computes a 64x64 tile.
// =====================================================================
__global__ __launch_bounds__(256) void proj_kernel(
    const float* __restrict__ Q, const float* __restrict__ K,
    const float* __restrict__ V, const float* __restrict__ Mq,
    const float* __restrict__ Mk, const float* __restrict__ Mv,
    _Float16* __restrict__ QdT, _Float16* __restrict__ KdT,
    _Float16* __restrict__ VdD) {
  __shared__ __align__(64) _Float16 Alds[64 * 32];   // weights (m,k)
  __shared__ __align__(64) _Float16 BldsT[64 * 32];  // activations (n,k)

  const int tid  = threadIdx.x;
  const int lane = tid & 31;
  const int wave = tid >> 5;
  const int n15  = lane & 15;
  const int hi   = lane >> 4;
  const int nbase = blockIdx.x * 64;

  const float* X; const float* W; _Float16* Y; float scale; bool dmajor;
  if (blockIdx.y == 0)      { X = Q; W = Mq; Y = QdT; scale = 1.0f;   dmajor = false; }
  else if (blockIdx.y == 1) { X = K; W = Mk; Y = KdT; scale = 0.125f; dmajor = false; }
  else                      { X = V; W = Mv; Y = VdD; scale = 1.0f;   dmajor = true;  }

  const int mt  = wave >> 1;
  const int nt0 = (wave & 1) * 2;

  v8f acc0 = {}, acc1 = {};

  for (int kc = 0; kc < D_IN; kc += 32) {
#pragma unroll
    for (int i = 0; i < 8; ++i) {               // weights (m,k), coalesced
      int e = tid * 8 + i;
      int m = e >> 5, k = e & 31;
      Alds[e] = (_Float16)W[m * D_IN + kc + k];
    }
#pragma unroll
    for (int i = 0; i < 8; ++i) {               // activations -> (n,k)-major
      int e = tid * 8 + i;
      int k = e >> 6, n = e & 63;
      BldsT[n * 32 + k] = (_Float16)X[(kc + k) * SEQ + nbase + n];
    }
    __syncthreads();

    v16h a  = load_afrag(&Alds[(mt * 16 + n15) * 32], hi);
    v16h b0 = *(const v16h*)&BldsT[((nt0    ) * 16 + n15) * 32 + 16 * hi];
    v16h b1 = *(const v16h*)&BldsT[((nt0 + 1) * 16 + n15) * 32 + 16 * hi];
    acc0 = wmma16(a, b0, acc0);
    acc1 = wmma16(a, b1, acc1);
    __syncthreads();
  }

  // Uniform layout branch hoisted OUT of the store loop (was generating
  // per-element cbranches).
  const int n0 = nbase + nt0 * 16 + n15;
  if (dmajor) {                                  // (d, seq): coalesced lanes
#pragma unroll
    for (int r = 0; r < 8; ++r) {
      int m = mt * 16 + r + 8 * hi;
      Y[(size_t)m * SEQ + n0     ] = (_Float16)(acc0[r] * scale);
      Y[(size_t)m * SEQ + n0 + 16] = (_Float16)(acc1[r] * scale);
    }
  } else {                                       // (seq, d)
#pragma unroll
    for (int r = 0; r < 8; ++r) {
      int m = mt * 16 + r + 8 * hi;
      Y[(size_t)(n0     ) * D_OUT + m] = (_Float16)(acc0[r] * scale);
      Y[(size_t)(n0 + 16) * D_OUT + m] = (_Float16)(acc1[r] * scale);
    }
  }
}

// =====================================================================
// Kernel 2: flash attention (swapped roles, transposed pipeline) with
// TDM double-buffered K/V staging when available.
// grid = SEQ/64, block 128 (4 waves); wave owns 16 t-columns.
// =====================================================================
__global__ __launch_bounds__(128) void flash_kernel(
    const _Float16* __restrict__ QdT,   // "keys"    (s,64)
    const _Float16* __restrict__ KdT,   // "queries" (t,64), prescaled
    const _Float16* __restrict__ VdD,   // "values"  (64,seq) d-major
    float* __restrict__ out) {          // (64, SEQ)
#ifdef HAVE_TDM
  __shared__ __align__(64) _Float16 Klds[2][64 * 64];   // (s_local, d)
  __shared__ __align__(64) _Float16 VldsT[2][64 * 64];  // (d, s_local)
#else
  __shared__ __align__(64) _Float16 Klds[1][64 * 64];
  __shared__ __align__(64) _Float16 VldsT[1][64 * 64];
#endif
  __shared__ __align__(64) _Float16 PldsT[4][16 * 64];  // per-wave (t, s_local)

  const int tid  = threadIdx.x;
  const int lane = tid & 31;
  const int wave = tid >> 5;
  const int n15  = lane & 15;
  const int hi   = lane >> 4;
  const int tbase = blockIdx.x * 64 + wave * 16;
  const int NB = SEQ / 64;

  // Persistent q^T B-fragments: one 32B global load per 32-d chunk.
  const v16h bq0 = *(const v16h*)&KdT[(size_t)(tbase + n15) * D_OUT + 16 * hi];
  const v16h bq1 = *(const v16h*)&KdT[(size_t)(tbase + n15) * D_OUT + 32 + 16 * hi];

  float m_t = -INFINITY, l_t = 0.0f;   // stats for t = tbase + n15
  v8f o[4] = {{}, {}, {}, {}};         // O' tiles (d x t)

  _Float16* Pl = &PldsT[wave][0];

#ifdef HAVE_TDM
  if (wave == 0) {                     // prime buffer 0 via the Tensor DMA
    tdm_load_2d(lds_off(&Klds[0][0]),  QdT,           64, SEQ, 64, 64, 64);
    tdm_load_2d(lds_off(&VldsT[0][0]), VdD,           SEQ, 64, 64, 64, SEQ);
  }
#endif

  for (int sb = 0; sb < NB; ++sb) {
    const int sbase = sb * 64;
#ifdef HAVE_TDM
    const int cur = sb & 1;
    if (wave == 0) {
      if (sb + 1 < NB) {               // kick off next block, then wait for
        const int nxt = cur ^ 1;       // current (2 newest ops outstanding)
        tdm_load_2d(lds_off(&Klds[nxt][0]),
                    QdT + (size_t)(sbase + 64) * D_OUT, 64, SEQ, 64, 64, 64);
        tdm_load_2d(lds_off(&VldsT[nxt][0]),
                    VdD + (sbase + 64),                 SEQ, 64, 64, 64, SEQ);
        __builtin_amdgcn_s_wait_tensorcnt(2);
      } else {
        __builtin_amdgcn_s_wait_tensorcnt(0);
      }
    }
    __syncthreads();                   // publish buffer `cur` to all waves
#else
    const int cur = 0;
    {  // manual staging fallback
      const uint4* srcK = (const uint4*)(QdT + (size_t)sbase * D_OUT);
      uint4* dK = (uint4*)&Klds[0][0];
#pragma unroll
      for (int i = 0; i < 4; ++i) dK[tid + i * 128] = srcK[tid + i * 128];
      const int vrow = tid >> 1, vcol = (tid & 1) * 32;
      const uint4* srcV = (const uint4*)(VdD + (size_t)vrow * SEQ + sbase + vcol);
      uint4* dV = (uint4*)(&VldsT[0][0] + vrow * 64 + vcol);
#pragma unroll
      for (int i = 0; i < 4; ++i) dV[i] = srcV[i];
    }
    if (sb + 1 < NB) {
      __builtin_prefetch(QdT + (size_t)(sbase + 64) * D_OUT + tid * 64, 0, 1);
      __builtin_prefetch(VdD + (size_t)(tid & 63) * SEQ + sbase + 64, 0, 1);
    }
    __syncthreads();
#endif

    // ---- S^T = Kblk x q^T : 4 tiles (16 s x 16 t), contraction d=64 ----
    v8f st[4];
#pragma unroll
    for (int stI = 0; stI < 4; ++stI) {
      const _Float16* krow = &Klds[cur][(stI * 16 + n15) * 64];
      v16h aK0 = load_afrag(krow, hi);
      v16h aK1 = load_afrag(krow + 32, hi);
      v8f s = {};
      s = wmma16(aK0, bq0, s);
      s = wmma16(aK1, bq1, s);
      st[stI] = s;
    }

    // ---- online softmax: per-lane stats for t = n15 ----
    float bm = -INFINITY;
#pragma unroll
    for (int stI = 0; stI < 4; ++stI)
#pragma unroll
      for (int r = 0; r < 8; ++r) bm = fmaxf(bm, st[stI][r]);
    bm = fmaxf(bm, __shfl_xor(bm, 16, 32));
    const float mnew  = fmaxf(m_t, bm);
    const float alpha = __expf(m_t - mnew);
    m_t = mnew;

    float ps = 0.0f;
#pragma unroll
    for (int stI = 0; stI < 4; ++stI) {
      v8h p;
#pragma unroll
      for (int r = 0; r < 8; ++r) {
        float pe = __expf(st[stI][r] - mnew);
        ps += pe;
        p[r] = (_Float16)pe;
      }
      *(v8h*)&Pl[n15 * 64 + stI * 16 + 8 * hi] = p;
    }
    ps += __shfl_xor(ps, 16, 32);
    l_t = l_t * alpha + ps;
#pragma unroll
    for (int dt = 0; dt < 4; ++dt) o[dt] = o[dt] * alpha;

    // ---- O' += V^T x P ----
    const v16h bp0 = *(const v16h*)&Pl[n15 * 64 + 16 * hi];
    const v16h bp1 = *(const v16h*)&Pl[n15 * 64 + 32 + 16 * hi];
#pragma unroll
    for (int dt = 0; dt < 4; ++dt) {
      const _Float16* vrow = &VldsT[cur][(dt * 16 + n15) * 64];
      v16h av0 = load_afrag(vrow, hi);
      v16h av1 = load_afrag(vrow + 32, hi);
      o[dt] = wmma16(av0, bp0, o[dt]);
      o[dt] = wmma16(av1, bp1, o[dt]);
    }
    __syncthreads();   // all reads of buffer `cur` done before it is refilled
  }

  // ---- epilogue: out[d*SEQ + t] = O'[d,t] / l_t ----
  const float inv_l = 1.0f / l_t;
  const int t = tbase + n15;
#pragma unroll
  for (int dt = 0; dt < 4; ++dt) {
#pragma unroll
    for (int r = 0; r < 8; ++r) {
      int d = dt * 16 + r + 8 * hi;
      out[(size_t)d * SEQ + t] = o[dt][r] * inv_l;
    }
  }
}

// =====================================================================
extern "C" void kernel_launch(void* const* d_in, const int* in_sizes, int n_in,
                              void* d_out, int out_size, void* d_ws,
                              size_t ws_size, hipStream_t stream) {
  const float* Q  = (const float*)d_in[0];
  const float* K  = (const float*)d_in[1];
  const float* V  = (const float*)d_in[2];
  const float* Mq = (const float*)d_in[3];
  const float* Mk = (const float*)d_in[4];
  const float* Mv = (const float*)d_in[5];
  float* out = (float*)d_out;

  _Float16* QdT = (_Float16*)d_ws;               // (SEQ,64) f16
  _Float16* KdT = QdT + (size_t)SEQ * D_OUT;     // (SEQ,64) f16, prescaled 1/8
  _Float16* VdD = KdT + (size_t)SEQ * D_OUT;     // (64,SEQ) f16

  dim3 pgrid(SEQ / 64, 3);
  proj_kernel<<<pgrid, 256, 0, stream>>>(Q, K, V, Mq, Mk, Mv, QdT, KdT, VdD);
  flash_kernel<<<SEQ / 64, 128, 0, stream>>>(QdT, KdT, VdD, out);
}
